// NER_Model_72069551227372
// MI455X (gfx1250) — compile-verified, matrix operand order checked by reading, not run on hardware
//
#include <hip/hip_runtime.h>

typedef __attribute__((ext_vector_type(16))) _Float16 v16h;
typedef __attribute__((ext_vector_type(8)))  float    v8f;

#define B_  1024
#define T_  512
#define L_  64
#define NEGV -10000.0f

#define LOG2E 1.4426950408889634f
#define LN2   0.6931471805599453f

// raw transcendentals: v_exp_f32 / v_log_f32 without denormal-guard fixup code
__device__ __forceinline__ float fast_exp(float x) {
    return __builtin_amdgcn_exp2f(x * LOG2E);
}
__device__ __forceinline__ float fast_log(float x) {
    return __builtin_amdgcn_logf(x) * LN2;
}

// ---------------------------------------------------------------------------
// Gold path score: gold[b] = sum_t trans[cur,prev] + feats[b,t+1,cur]
// One wave32 per batch row; lanes stride over t; wave reduction.
// ---------------------------------------------------------------------------
__global__ __launch_bounds__(256) void crf_gold_kernel(
    const float* __restrict__ feats,
    const float* __restrict__ trans,
    const int*   __restrict__ labels,
    float*       __restrict__ gold)
{
    const int lane = threadIdx.x & 31;
    const int wave = threadIdx.x >> 5;
    const int b    = blockIdx.x * 8 + wave;      // 128 blocks * 8 waves = 1024

    const int*   lb = labels + (size_t)b * T_;
    const float* fb = feats  + (size_t)b * T_ * L_;

    float s = 0.0f;
    for (int t = 1 + lane; t < T_; t += 32) {
        const int cur  = lb[t];
        const int prev = lb[t - 1];
        s += trans[cur * L_ + prev] + fb[t * L_ + cur];
    }
    #pragma unroll
    for (int off = 16; off > 0; off >>= 1)
        s += __shfl_xor(s, off);
    if (lane == 0) gold[b] = s;
}

// ---------------------------------------------------------------------------
// CRF forward scan on WMMA.
//   step:  alpha'[b,i] = m[b] + log( sum_j exp(trans[i,j]) * exp(alpha[b,j]-m[b]) )
//                        + feats[b,t,i]
// GEMM orientation: D[i, b] = A[i, j] * B[j, b]
//   A = exp(trans)  : constant -> 4 (M-tiles) x 2 (K-tiles) v16h frags, loaded once
//   B = exp(alpha-m): rebuilt per step, PURELY per-lane from the D-layout alpha
// Fragment index math (wave32, from ISA 7.12.2):
//   A frag [mi][kt], lane L, elem e: i = 16*mi + (L%16),
//                                    j = 32*kt + 16*(e/8) + 8*(L/16) + (e%8)
//   B frag [kt],     lane L, elem e: b = (L%16),  j = same formula as A's j
//   D frag [mi],     lane L, elem r: i = 16*mi + r + 8*(L/16), b = (L%16)
// => B[kt][e] corresponds to alpha[mi = 2*kt + e/8][r = e%8] on the SAME lane:
//    the per-step "transpose" is a pure per-lane register remap.
// Feats loads for step t+1 are issued before step t's WMMA/log chain
// (software pipelining); they depend only on t, not on the recurrence.
// One wave handles 16 batch rows; 64 blocks of 32 threads cover B=1024.
// ---------------------------------------------------------------------------
__global__ __launch_bounds__(32) void crf_forward_kernel(
    const float* __restrict__ feats,
    const float* __restrict__ trans,
    const float* __restrict__ gold,
    float*       __restrict__ out)
{
    const int lane = threadIdx.x;
    const int n    = lane & 15;     // batch column within tile
    const int g    = lane >> 4;     // lane group
    const int b    = blockIdx.x * 16 + n;

    // --- constant A = exp(transitions), loaded once into registers ---------
    v16h A[4][2];
    #pragma unroll
    for (int mi = 0; mi < 4; ++mi) {
        const int i = 16 * mi + n;
        #pragma unroll
        for (int kt = 0; kt < 2; ++kt) {
            #pragma unroll
            for (int e = 0; e < 16; ++e) {
                const int j = 32 * kt + 16 * (e >> 3) + 8 * g + (e & 7);
                A[mi][kt][e] = (_Float16)fast_exp(trans[i * L_ + j]);
            }
        }
    }

    // --- alpha state in D-layout: alpha[mi][r] <-> (b, i=16*mi+r+8*g) ------
    v8f alpha[4];
    #pragma unroll
    for (int mi = 0; mi < 4; ++mi)
        #pragma unroll
        for (int r = 0; r < 8; ++r)
            alpha[mi][r] = (16 * mi + r + 8 * g == 0) ? 0.0f : NEGV;  // START=0

    const float* fb = feats + (size_t)b * T_ * L_;

    // prologue: preload feats for t=1
    float4 f0[4], f1[4];
    #pragma unroll
    for (int mi = 0; mi < 4; ++mi) {
        f0[mi] = *reinterpret_cast<const float4*>(fb + 1 * L_ + 16 * mi + 8 * g);
        f1[mi] = *reinterpret_cast<const float4*>(fb + 1 * L_ + 16 * mi + 8 * g + 4);
    }

    for (int t = 1; t < T_; ++t) {
        // row max over i (per batch b): per-lane max, then combine lane pair
        float m = alpha[0][0];
        #pragma unroll
        for (int mi = 0; mi < 4; ++mi)
            #pragma unroll
            for (int r = 0; r < 8; ++r)
                m = fmaxf(m, alpha[mi][r]);
        m = fmaxf(m, __shfl_xor(m, 16));

        // B = exp(alpha - m): per-lane register remap into B-fragment layout
        v16h Bf[2];
        #pragma unroll
        for (int kt = 0; kt < 2; ++kt)
            #pragma unroll
            for (int e = 0; e < 16; ++e)
                Bf[kt][e] = (_Float16)fast_exp(alpha[2 * kt + (e >> 3)][e & 7] - m);

        // issue next step's feats loads now (independent of the WMMA chain);
        // clamp keeps the access in-bounds and the branch-free select keeps
        // EXEC uniform (required for WMMA)
        const int tn = (t + 1 < T_) ? (t + 1) : t;
        const float* frn = fb + tn * L_;
        float4 nf0[4], nf1[4];
        #pragma unroll
        for (int mi = 0; mi < 4; ++mi) {
            nf0[mi] = *reinterpret_cast<const float4*>(frn + 16 * mi + 8 * g);
            nf1[mi] = *reinterpret_cast<const float4*>(frn + 16 * mi + 8 * g + 4);
        }

        // GEMM: acc[i,b] = sum_j exp(trans[i,j]) * exp(alpha[b,j]-m[b])
        v8f acc[4] = {};
        #pragma unroll
        for (int mi = 0; mi < 4; ++mi) {
            acc[mi] = __builtin_amdgcn_wmma_f32_16x16x32_f16(
                false, A[mi][0], false, Bf[0], (short)0, acc[mi], false, false);
            acc[mi] = __builtin_amdgcn_wmma_f32_16x16x32_f16(
                false, A[mi][1], false, Bf[1], (short)0, acc[mi], false, false);
        }

        // alpha' = m + log(acc) + feats[b,t,i]   (i = 16*mi + r + 8*g)
        #pragma unroll
        for (int mi = 0; mi < 4; ++mi) {
            alpha[mi][0] = m + fast_log(acc[mi][0]) + f0[mi].x;
            alpha[mi][1] = m + fast_log(acc[mi][1]) + f0[mi].y;
            alpha[mi][2] = m + fast_log(acc[mi][2]) + f0[mi].z;
            alpha[mi][3] = m + fast_log(acc[mi][3]) + f0[mi].w;
            alpha[mi][4] = m + fast_log(acc[mi][4]) + f1[mi].x;
            alpha[mi][5] = m + fast_log(acc[mi][5]) + f1[mi].y;
            alpha[mi][6] = m + fast_log(acc[mi][6]) + f1[mi].z;
            alpha[mi][7] = m + fast_log(acc[mi][7]) + f1[mi].w;
            f0[mi] = nf0[mi];
            f1[mi] = nf1[mi];
        }
    }

    // logZ[b] = logsumexp_i alpha[b,i]
    float m = alpha[0][0];
    #pragma unroll
    for (int mi = 0; mi < 4; ++mi)
        #pragma unroll
        for (int r = 0; r < 8; ++r)
            m = fmaxf(m, alpha[mi][r]);
    m = fmaxf(m, __shfl_xor(m, 16));

    float s = 0.0f;
    #pragma unroll
    for (int mi = 0; mi < 4; ++mi)
        #pragma unroll
        for (int r = 0; r < 8; ++r)
            s += fast_exp(alpha[mi][r] - m);
    s += __shfl_xor(s, 16);

    const float logZ = m + fast_log(s);
    if (lane < 16) out[b] = logZ - gold[b];
}

// ---------------------------------------------------------------------------
extern "C" void kernel_launch(void* const* d_in, const int* in_sizes, int n_in,
                              void* d_out, int out_size, void* d_ws, size_t ws_size,
                              hipStream_t stream)
{
    const float* feats  = (const float*)d_in[0];   // (B,T,L) f32
    const float* trans  = (const float*)d_in[1];   // (L,L)   f32
    const int*   labels = (const int*)d_in[2];     // (B,T)   int
    float*       out    = (float*)d_out;           // (B,)    f32
    float*       gold   = (float*)d_ws;            // B floats of scratch

    crf_gold_kernel<<<B_ / 8, 256, 0, stream>>>(feats, trans, labels, gold);
    crf_forward_kernel<<<B_ / 16, 32, 0, stream>>>(feats, trans, gold, out);
}
